// Qwen3ExplicitLMBlock_54949811585329
// MI455X (gfx1250) — compile-verified, matrix operand order checked by reading, not run on hardware
//
#include <hip/hip_runtime.h>
#include <cstdint>
#include <cstddef>

// ---------------- types ----------------
typedef __bf16 bf16;
typedef __attribute__((ext_vector_type(16))) __bf16 v16bf;
typedef __attribute__((ext_vector_type(8)))  float  v8f;

union FragBF { uint4 u[2]; v16bf v; };

#if __has_builtin(__builtin_amdgcn_global_load_async_to_lds_b128) && \
    __has_builtin(__builtin_amdgcn_s_wait_asynccnt)
#define USE_ASYNC_LDS 1
typedef int v4i_async __attribute__((vector_size(16)));
typedef __attribute__((address_space(1))) v4i_async* gv4i_p;
typedef __attribute__((address_space(3))) v4i_async* lv4i_p;
#else
#define USE_ASYNC_LDS 0
#endif

// ---------------- problem constants ----------------
constexpr int cB = 2, cS = 1024, cD = 1024, cH = 16, cKV = 8, cHD = 128, cFFN = 3072;
constexpr int cNKEYS = 256, cKD = 128, cNC = 8, cKNUM = cNKEYS * cNKEYS, cL = 8;
constexpr int cM = cB * cS;               // 2048 rows
constexpr int OUT_ELEMS = cM * cD;        // 2097152
constexpr int SIM_OFF = OUT_ELEMS;
constexpr int DIV_OFF = OUT_ELEMS + 1;
constexpr int CNT_OFF = OUT_ELEMS + 2;

// =====================================================================
// Weight prep: transpose f32 [K,N] -> bf16 [N,K]
// =====================================================================
__global__ void transpose_bf16_kernel(const float* __restrict__ W, bf16* __restrict__ Wt,
                                      int K, int N) {
  int i = blockIdx.x * 256 + threadIdx.x;
  if (i >= K * N) return;
  int k = i / N, n = i - k * N;
  Wt[(size_t)n * K + k] = (bf16)W[i];
}

__global__ void cvt_f32_bf16_kernel(const float* __restrict__ a, bf16* __restrict__ b, int n) {
  int i = blockIdx.x * 256 + threadIdx.x;
  if (i < n) b[i] = (bf16)a[i];
}

// =====================================================================
// RMSNorm: x f32 [rows, Dm] -> yb bf16 (stride ldyb), optional yf f32
// =====================================================================
__global__ __launch_bounds__(256)
void rmsnorm_kernel(const float* __restrict__ x, const float* __restrict__ w,
                    float* __restrict__ yf, bf16* __restrict__ yb, int ldyb, int Dm) {
  __shared__ float red[256];
  int row = blockIdx.x;
  const float* xr = x + (size_t)row * Dm;
  float ss = 0.f;
  for (int i = threadIdx.x; i < Dm; i += 256) { float v = xr[i]; ss += v * v; }
  red[threadIdx.x] = ss; __syncthreads();
  for (int s = 128; s > 0; s >>= 1) { if (threadIdx.x < s) red[threadIdx.x] += red[threadIdx.x + s]; __syncthreads(); }
  float inv = rsqrtf(red[0] / (float)Dm + 1e-6f);
  for (int i = threadIdx.x; i < Dm; i += 256) {
    float v = xr[i] * inv * w[i];
    if (yf) yf[(size_t)row * Dm + i] = v;
    yb[(size_t)row * ldyb + i] = (bf16)v;
  }
}

// =====================================================================
// Generic bf16 WMMA GEMM: C[M,N] = A[M,K] * Wt[N,K]^T  (f32 accum/out)
// grid = (N/256, M/64); block = 256 (8 waves); wave tile 32x64
// Ping-pong double buffering (K must be a multiple of 64): no register
// rotation copies, loads for one buffer overlap WMMAs of the other.
// =====================================================================
__global__ __launch_bounds__(256)
void gemm_bf16_kernel(const bf16* __restrict__ A, int lda,
                      const bf16* __restrict__ Bt, int ldb,
                      float* __restrict__ C, int ldc, int Kdim) {
  const int wave = threadIdx.x >> 5;
  const int lane = threadIdx.x & 31;
  const int lr = lane & 15, lh = lane >> 4;
  const int mb = blockIdx.y * 64 + (wave & 1) * 32;
  const int nb = blockIdx.x * 256 + (wave >> 1) * 64;
  v8f acc[2][4] = {};
  const bf16* aptr0 = A + (size_t)(mb + lr) * lda + lh * 8;
  const bf16* aptr1 = A + (size_t)(mb + 16 + lr) * lda + lh * 8;
  const bf16* bbase = Bt + (size_t)(nb + lr) * ldb + lh * 8;

  auto loadA = [&](FragBF* af, int k0) {
    af[0].u[0] = *(const uint4*)(aptr0 + k0);
    af[0].u[1] = *(const uint4*)(aptr0 + k0 + 16);
    af[1].u[0] = *(const uint4*)(aptr1 + k0);
    af[1].u[1] = *(const uint4*)(aptr1 + k0 + 16);
  };
  auto loadB = [&](FragBF* bfr, int k0) {
#pragma unroll
    for (int j = 0; j < 4; ++j) {
      const bf16* bp = bbase + (size_t)j * 16 * ldb + k0;
      bfr[j].u[0] = *(const uint4*)(bp);
      bfr[j].u[1] = *(const uint4*)(bp + 16);
    }
  };

  FragBF afA[2], bfA[4], afB[2], bfB[4];
  loadA(afA, 0);
  loadB(bfA, 0);
  for (int k0 = 0; k0 < Kdim; k0 += 64) {
    // load B-set for k0+32 while A-set WMMAs issue
    loadA(afB, k0 + 32);
    loadB(bfB, k0 + 32);
    __builtin_prefetch(aptr0 + k0 + 256, 0, 0);
#pragma unroll
    for (int j = 0; j < 4; ++j) {
      acc[0][j] = __builtin_amdgcn_wmma_f32_16x16x32_bf16(false, afA[0].v, false, bfA[j].v,
                                                          (short)0, acc[0][j], false, false);
      acc[1][j] = __builtin_amdgcn_wmma_f32_16x16x32_bf16(false, afA[1].v, false, bfA[j].v,
                                                          (short)0, acc[1][j], false, false);
    }
    // reload A-set for k0+64 while B-set WMMAs issue
    if (k0 + 64 < Kdim) {
      loadA(afA, k0 + 64);
      loadB(bfA, k0 + 64);
    }
#pragma unroll
    for (int j = 0; j < 4; ++j) {
      acc[0][j] = __builtin_amdgcn_wmma_f32_16x16x32_bf16(false, afB[0].v, false, bfB[j].v,
                                                          (short)0, acc[0][j], false, false);
      acc[1][j] = __builtin_amdgcn_wmma_f32_16x16x32_bf16(false, afB[1].v, false, bfB[j].v,
                                                          (short)0, acc[1][j], false, false);
    }
  }
#pragma unroll
  for (int mi = 0; mi < 2; ++mi)
#pragma unroll
    for (int j = 0; j < 4; ++j)
#pragma unroll
      for (int r = 0; r < 8; ++r)
        C[(size_t)(mb + mi * 16 + r + 8 * lh) * ldc + (nb + j * 16 + lr)] = acc[mi][j][r];
}

// =====================================================================
// Per-head RMSNorm + RoPE, f32 in -> bf16 out. grid=(M, nheads), block=HD
// =====================================================================
__global__ __launch_bounds__(128)
void qknorm_rope_kernel(const float* __restrict__ X, const float* __restrict__ wn,
                        bf16* __restrict__ Y, int nheads) {
  __shared__ float red[128];
  __shared__ float xs[128];
  int m = blockIdx.x, h = blockIdx.y, d = threadIdx.x;
  int pos = m & (cS - 1);
  const float* xr = X + ((size_t)m * nheads + h) * cHD;
  float v = xr[d];
  xs[d] = v;
  red[d] = v * v; __syncthreads();
  for (int t = 64; t > 0; t >>= 1) { if (d < t) red[d] += red[d + t]; __syncthreads(); }
  float inv = rsqrtf(red[0] / (float)cHD + 1e-6f);
  int dp = (d < 64) ? d + 64 : d - 64;
  float xv = v * inv * wn[d];
  float xp = xs[dp] * inv * wn[dp];
  float invfreq = powf(1.0e6f, -((float)(d & 63)) / 64.0f);
  float ang = (float)pos * invfreq;
  float rot = (d < 64) ? -xp : xp;
  float out = xv * __cosf(ang) + rot * __sinf(ang);
  Y[((size_t)m * nheads + h) * cHD + d] = (bf16)out;
}

// =====================================================================
// Flash attention (causal, GQA 2:1). One wave per 16-query tile.
// grid = (S/16, H, B), block = 32.
// V tile staged to LDS with async global->LDS copies when available.
// =====================================================================
__global__ __launch_bounds__(32)
void flash_attn_kernel(const bf16* __restrict__ Qm, const bf16* __restrict__ Km,
                       const bf16* __restrict__ Vm, bf16* __restrict__ Om) {
  __shared__ bf16 ldsP[16 * 32];
  __shared__ bf16 ldsV[32 * 128];    // [k][d] row-major
  const int qb = blockIdx.x, h = blockIdx.y, b = blockIdx.z;
  const int kvh = h >> 1;            // H/KV = 2
  const int lane = threadIdx.x & 31;
  const int lr = lane & 15, lh = lane >> 4;
  const float scale = 0.08838834764831845f;  // 128^-0.5

  FragBF aq[4];
  {
    const bf16* qp = Qm + ((size_t)(b * cS + qb * 16 + lr) * (cH * cHD)) + h * cHD + lh * 8;
#pragma unroll
    for (int c = 0; c < 4; ++c) {
      aq[c].u[0] = *(const uint4*)(qp + c * 32);
      aq[c].u[1] = *(const uint4*)(qp + c * 32 + 16);
    }
  }
  v8f oacc[8] = {};
  float mrow[8], lrow[8];
#pragma unroll
  for (int r = 0; r < 8; ++r) { mrow[r] = -3.0e38f; lrow[r] = 0.f; }

  const int nkb = (qb * 16 + 15) / 32 + 1;
  for (int kb = 0; kb < nkb; ++kb) {
    // -------- stage V tile (32 x 128) into LDS, lane owns key-row `lane` --
    {
      const bf16* vg = Vm + ((size_t)(b * cS + kb * 32 + lane) * (cKV * cHD)) + kvh * cHD;
      bf16* lrowp = ldsV + lane * 128;
#if USE_ASYNC_LDS
#pragma unroll
      for (int i = 0; i < 16; ++i)
        __builtin_amdgcn_global_load_async_to_lds_b128((gv4i_p)(vg + i * 8),
                                                       (lv4i_p)(lrowp + i * 8), 0, 0);
#else
#pragma unroll
      for (int i = 0; i < 16; ++i)
        *(uint4*)(lrowp + i * 8) = *(const uint4*)(vg + i * 8);
#endif
    }
    // -------- scores: two 16x16 WMMA tiles over HD=128 --------
    v8f st[2] = {};
#pragma unroll
    for (int t = 0; t < 2; ++t) {
      const bf16* kp = Km + ((size_t)(b * cS + kb * 32 + t * 16 + lr) * (cKV * cHD)) + kvh * cHD + lh * 8;
#pragma unroll
      for (int c = 0; c < 4; ++c) {
        FragBF bk;
        bk.u[0] = *(const uint4*)(kp + c * 32);
        bk.u[1] = *(const uint4*)(kp + c * 32 + 16);
        st[t] = __builtin_amdgcn_wmma_f32_16x16x32_bf16(false, aq[c].v, false, bk.v,
                                                        (short)0, st[t], false, false);
      }
    }
    // -------- online softmax update --------
#pragma unroll
    for (int r = 0; r < 8; ++r) {
      int qrow = qb * 16 + r + 8 * lh;
      float s0 = st[0][r] * scale;
      float s1 = st[1][r] * scale;
      if (kb * 32 + lr > qrow)      s0 = -1.0e30f;
      if (kb * 32 + 16 + lr > qrow) s1 = -1.0e30f;
      float mx = fmaxf(s0, s1);
#pragma unroll
      for (int o = 1; o < 16; o <<= 1) mx = fmaxf(mx, __shfl_xor(mx, o, 32));
      float mnew = fmaxf(mrow[r], mx);
      float p0 = __expf(s0 - mnew);
      float p1 = __expf(s1 - mnew);
      float sum = p0 + p1;
#pragma unroll
      for (int o = 1; o < 16; o <<= 1) sum += __shfl_xor(sum, o, 32);
      float alpha = __expf(mrow[r] - mnew);
      lrow[r] = lrow[r] * alpha + sum;
      mrow[r] = mnew;
      int row = r + 8 * lh;
      ldsP[row * 32 + lr]      = (bf16)p0;
      ldsP[row * 32 + 16 + lr] = (bf16)p1;
#pragma unroll
      for (int j = 0; j < 8; ++j) oacc[j][r] *= alpha;
    }
#if USE_ASYNC_LDS
    __builtin_amdgcn_s_wait_asynccnt(0);
#endif
    __syncthreads();
    // -------- O += P x V --------
    FragBF ap;
    ap.u[0] = *(const uint4*)&ldsP[lr * 32 + lh * 8];
    ap.u[1] = *(const uint4*)&ldsP[lr * 32 + lh * 8 + 16];
#pragma unroll
    for (int j = 0; j < 8; ++j) {
      FragBF bv;
#pragma unroll
      for (int e = 0; e < 8; ++e) {
        int k = lh * 8 + e;
        bv.v[e]     = ldsV[k * 128 + j * 16 + lr];
        bv.v[e + 8] = ldsV[(k + 16) * 128 + j * 16 + lr];
      }
      oacc[j] = __builtin_amdgcn_wmma_f32_16x16x32_bf16(false, ap.v, false, bv.v,
                                                        (short)0, oacc[j], false, false);
    }
    __syncthreads();
  }
  // -------- normalize + store bf16 --------
#pragma unroll
  for (int j = 0; j < 8; ++j)
#pragma unroll
    for (int r = 0; r < 8; ++r) {
      float v = oacc[j][r] / lrow[r];
      Om[((size_t)(b * cS + qb * 16 + r + 8 * lh) * (cH * cHD)) + h * cHD + j * 16 + lr] = (bf16)v;
    }
}

// =====================================================================
// simple elementwise kernels
// =====================================================================
__global__ void add_kernel(const float* __restrict__ a, const float* __restrict__ b,
                           float* __restrict__ c, int n) {
  int i = blockIdx.x * 256 + threadIdx.x;
  if (i < n) c[i] = a[i] + b[i];
}

__global__ void silu_mul_kernel(const float* __restrict__ g, const float* __restrict__ u,
                                bf16* __restrict__ o, int n) {
  int i = blockIdx.x * 256 + threadIdx.x;
  if (i < n) {
    float gv = g[i];
    float s = gv / (1.f + __expf(-gv));
    o[i] = (bf16)(s * u[i]);
  }
}

__global__ void fuse_out_kernel(const float* __restrict__ hs, const float* __restrict__ gatelin,
                                const float* __restrict__ memlin, float* __restrict__ out, int n) {
  int i = blockIdx.x * 256 + threadIdx.x;
  if (i < n) {
    float g = 1.f / (1.f + __expf(-gatelin[i]));
    out[i] = hs[i] + g * memlin[i];
  }
}

__global__ void zero_tail_kernel(float* __restrict__ out) {
  int i = blockIdx.x * 256 + threadIdx.x;
  if (i < 2 + cKNUM) out[OUT_ELEMS + i] = 0.f;
}

// =====================================================================
// product-key top-8 selection (per row, serial — trivially cheap)
// =====================================================================
__device__ void topk8(const float* a, int n, float* v, int* id) {
  for (int c = 0; c < 8; ++c) {
    float bv = -3.4e38f; int bi = 0;
    for (int i = 0; i < n; ++i) {
      bool used = false;
      for (int p = 0; p < c; ++p) used = used || (id[p] == i);
      float x = a[i];
      if (!used && x > bv) { bv = x; bi = i; }
    }
    v[c] = bv; id[c] = bi;
  }
}

__global__ void topk_select_kernel(const float* __restrict__ s1, const float* __restrict__ s2,
                                   int* __restrict__ idx_out) {
  int m = blockIdx.x * blockDim.x + threadIdx.x;
  if (m >= cM) return;
  float v1[8], v2[8]; int i1[8], i2[8];
  topk8(s1 + (size_t)m * cNKEYS, cNKEYS, v1, i1);
  topk8(s2 + (size_t)m * cNKEYS, cNKEYS, v2, i2);
  float comb[64];
  for (int i = 0; i < 8; ++i)
    for (int j = 0; j < 8; ++j) comb[i * 8 + j] = v1[i] + v2[j];
  float cv[8]; int ci[8];
  topk8(comb, 64, cv, ci);
  for (int c = 0; c < 8; ++c)
    idx_out[(size_t)m * 8 + c] = i1[ci[c] / 8] * cNKEYS + i2[ci[c] % 8];
}

// =====================================================================
// gather tokens + mean embed: mem[mc, :] over grid of M*NC blocks
// =====================================================================
__global__ __launch_bounds__(256)
void gather_mem_kernel(const int* __restrict__ idx, const int* __restrict__ bank,
                       const float* __restrict__ emb, float* __restrict__ mem) {
  int mc = blockIdx.x;
  int key = idx[mc];
  const int* toks = bank + (size_t)key * cL;
  int t[8];
#pragma unroll
  for (int l = 0; l < 8; ++l) t[l] = toks[l];
  for (int d = threadIdx.x; d < cD; d += 256) {
    float acc = 0.f;
#pragma unroll
    for (int l = 0; l < 8; ++l) acc += emb[(size_t)t[l] * cD + d];
    mem[(size_t)mc * cD + d] = acc * 0.125f;
  }
}

// =====================================================================
// cosine similarity h vs mem candidates; also raw mem norms
// =====================================================================
__global__ __launch_bounds__(256)
void cos_sim_kernel(const float* __restrict__ h, const float* __restrict__ mem,
                    float* __restrict__ sim, float* __restrict__ mnorm) {
  __shared__ float rhm[256], rmm[256], rhh[256];
  int mc = blockIdx.x, m = mc >> 3;
  const float* hr = h + (size_t)m * cD;
  const float* mr = mem + (size_t)mc * cD;
  float hm = 0.f, mm = 0.f, hh = 0.f;
  for (int i = threadIdx.x; i < cD; i += 256) {
    float hv = hr[i], mv = mr[i];
    hm += hv * mv; mm += mv * mv; hh += hv * hv;
  }
  rhm[threadIdx.x] = hm; rmm[threadIdx.x] = mm; rhh[threadIdx.x] = hh;
  __syncthreads();
  for (int s = 128; s > 0; s >>= 1) {
    if (threadIdx.x < s) {
      rhm[threadIdx.x] += rhm[threadIdx.x + s];
      rmm[threadIdx.x] += rmm[threadIdx.x + s];
      rhh[threadIdx.x] += rhh[threadIdx.x + s];
    }
    __syncthreads();
  }
  if (threadIdx.x == 0) {
    float hn = fmaxf(sqrtf(rhh[0]), 1e-8f);
    float mnr = sqrtf(rmm[0]);
    float mn = fmaxf(mnr, 1e-8f);
    sim[mc] = rhm[0] / (hn * mn);
    mnorm[mc] = mnr;
  }
}

// =====================================================================
// diversity loss: mean off-diagonal cosine among the 8 candidates
// grid = (M, 28 pairs), block = 256; atomicAdd scaled contribution
// =====================================================================
__global__ __launch_bounds__(256)
void div_loss_kernel(const float* __restrict__ mem, const float* __restrict__ mnorm,
                     float* __restrict__ div_acc) {
  __shared__ float red[256];
  int m = blockIdx.x;
  int p = blockIdx.y, i = 0, t = p;
  while (t >= 7 - i) { t -= 7 - i; ++i; }
  int j = i + 1 + t;
  const float* a = mem + ((size_t)m * 8 + i) * cD;
  const float* b = mem + ((size_t)m * 8 + j) * cD;
  float d = 0.f;
  for (int k = threadIdx.x; k < cD; k += 256) d += a[k] * b[k];
  red[threadIdx.x] = d; __syncthreads();
  for (int s = 128; s > 0; s >>= 1) { if (threadIdx.x < s) red[threadIdx.x] += red[threadIdx.x + s]; __syncthreads(); }
  if (threadIdx.x == 0) {
    float ni = fmaxf(mnorm[m * 8 + i], 1e-12f);
    float nj = fmaxf(mnorm[m * 8 + j], 1e-12f);
    float val = red[0] / (ni * nj);
    atomicAdd(div_acc, 2.0f * val / (float)(cM * cNC * (cNC - 1)));
  }
}

// =====================================================================
// gumbel-softmax straight-through select + sim loss + scatter counts
// =====================================================================
__global__ void select_gumbel_kernel(const float* __restrict__ sim, const float* __restrict__ gu,
                                     const int* __restrict__ idx, float* __restrict__ selw,
                                     float* __restrict__ simloss, float* __restrict__ counts) {
  int m = blockIdx.x * blockDim.x + threadIdx.x;
  if (m >= cM) return;
  float s[8], z[8], e[8];
  float mx = -3.4e38f;
#pragma unroll
  for (int c = 0; c < 8; ++c) {
    s[c] = sim[m * 8 + c];
    float u = gu[m * 8 + c];
    float g = -__logf(-__logf(u + 1e-20f) + 1e-20f);
    z[c] = s[c] + g;
    mx = fmaxf(mx, z[c]);
  }
  float sum = 0.f;
#pragma unroll
  for (int c = 0; c < 8; ++c) { e[c] = __expf(z[c] - mx); sum += e[c]; }
  int am = 0; float bm = z[0];
#pragma unroll
  for (int c = 1; c < 8; ++c) if (z[c] > bm) { bm = z[c]; am = c; }
  float selsim = 0.f;
#pragma unroll
  for (int c = 0; c < 8; ++c) {
    float soft = e[c] / sum;
    float hard = (c == am) ? 1.f : 0.f;
    float w = (hard - soft) + soft;   // straight-through, same arithmetic as ref
    selw[m * 8 + c] = w;
    selsim += s[c] * w;
    atomicAdd(&counts[idx[m * 8 + c]], w);
  }
  atomicAdd(simloss, -selsim / (float)cM);
}

// =====================================================================
// weighted candidate sum -> bf16 second half of cat buffer
// =====================================================================
__global__ __launch_bounds__(256)
void selmem_kernel(const float* __restrict__ mem, const float* __restrict__ selw,
                   bf16* __restrict__ cat2 /* stride 2048, offset +1024 applied */) {
  int m = blockIdx.x;
  float w[8];
#pragma unroll
  for (int c = 0; c < 8; ++c) w[c] = selw[m * 8 + c];
  for (int d = threadIdx.x; d < cD; d += 256) {
    float acc = 0.f;
#pragma unroll
    for (int c = 0; c < 8; ++c) acc += w[c] * mem[((size_t)m * 8 + c) * cD + d];
    cat2[(size_t)m * (2 * cD) + d] = (bf16)acc;
  }
}

// =====================================================================
// host orchestration
// =====================================================================
static inline int cdiv(int a, int b) { return (a + b - 1) / b; }

extern "C" void kernel_launch(void* const* d_in, const int* in_sizes, int n_in,
                              void* d_out, int out_size, void* d_ws, size_t ws_size,
                              hipStream_t stream) {
  (void)in_sizes; (void)n_in; (void)out_size; (void)ws_size;
  const float* x_in   = (const float*)d_in[0];
  const int*   bank   = (const int*)d_in[1];
  const float* gumbel = (const float*)d_in[2];
  const float* wq  = (const float*)d_in[3];
  const float* wk  = (const float*)d_in[4];
  const float* wv  = (const float*)d_in[5];
  const float* wo  = (const float*)d_in[6];
  const float* qn  = (const float*)d_in[7];
  const float* kn  = (const float*)d_in[8];
  const float* ln1 = (const float*)d_in[9];
  const float* ln2 = (const float*)d_in[10];
  const float* memn = (const float*)d_in[11];
  const float* wg  = (const float*)d_in[12];
  const float* wu  = (const float*)d_in[13];
  const float* wd  = (const float*)d_in[14];
  const float* wqr = (const float*)d_in[15];
  const float* pkk = (const float*)d_in[16];
  const float* wfg = (const float*)d_in[17];
  const float* wfm = (const float*)d_in[18];
  const float* temb = (const float*)d_in[19];

  float* out = (float*)d_out;

  // ---- workspace layout ----
  char* base = (char*)d_ws;
  size_t off = 0;
  auto take = [&](size_t bytes) -> void* {
    void* p = (void*)(base + off);
    off += (bytes + 255) & ~(size_t)255;
    return p;
  };
  bf16* wq_t  = (bf16*)take((size_t)2048 * 1024 * 2);
  bf16* wk_t  = (bf16*)take((size_t)1024 * 1024 * 2);
  bf16* wv_t  = (bf16*)take((size_t)1024 * 1024 * 2);
  bf16* wo_t  = (bf16*)take((size_t)1024 * 2048 * 2);
  bf16* wg_t  = (bf16*)take((size_t)3072 * 1024 * 2);
  bf16* wu_t  = (bf16*)take((size_t)3072 * 1024 * 2);
  bf16* wd_t  = (bf16*)take((size_t)1024 * 3072 * 2);
  bf16* wqr_t = (bf16*)take((size_t)256 * 1024 * 2);
  bf16* pk_t  = (bf16*)take((size_t)2 * 256 * 128 * 2);
  bf16* wfg_t = (bf16*)take((size_t)1024 * 2048 * 2);
  bf16* wfm_t = (bf16*)take((size_t)1024 * 1024 * 2);

  bf16*  h1_bf  = (bf16*)take((size_t)cM * 1024 * 2);
  float* x1_f   = (float*)take((size_t)cM * 1024 * 4);
  bf16*  h2_bf  = (bf16*)take((size_t)cM * 1024 * 2);
  float* hs_f   = (float*)take((size_t)cM * 1024 * 4);
  float* hmem_f = (float*)take((size_t)cM * 1024 * 4);
  bf16*  cat_bf = (bf16*)take((size_t)cM * 2048 * 2);
  float* gatelin = (float*)take((size_t)cM * 1024 * 4);
  float* memlin  = (float*)take((size_t)cM * 1024 * 4);
  int*   idx    = (int*)take((size_t)cM * 8 * 4);
  float* selw   = (float*)take((size_t)cM * 8 * 4);
  float* sim    = (float*)take((size_t)cM * 8 * 4);
  float* mnorm  = (float*)take((size_t)cM * 8 * 4);

  char* arena = (char*)take((size_t)80 << 20);
  // phase 1 (attention)
  float* q_f    = (float*)(arena);
  float* k_f    = (float*)(arena + ((size_t)16 << 20));
  float* v_f    = (float*)(arena + ((size_t)24 << 20));
  bf16*  q_bf   = (bf16*)(arena + ((size_t)32 << 20));
  bf16*  k_bf   = (bf16*)(arena + ((size_t)40 << 20));
  bf16*  v_bf   = (bf16*)(arena + ((size_t)44 << 20));
  bf16*  attn_bf = (bf16*)(arena + ((size_t)48 << 20));
  float* oproj  = (float*)(arena + ((size_t)56 << 20));
  // phase 2 (MLP)
  float* gate_f = (float*)(arena);
  float* up_f   = (float*)(arena + ((size_t)24 << 20));
  bf16*  act_bf = (bf16*)(arena + ((size_t)48 << 20));
  float* mlp_f  = (float*)(arena + ((size_t)60 << 20));
  // phase 3 (memory)
  float* qpk_f  = (float*)(arena);
  bf16*  qpk_bf = (bf16*)(arena + ((size_t)2 << 20));
  float* s1_f   = (float*)(arena + ((size_t)3 << 20));
  float* s2_f   = (float*)(arena + ((size_t)5 << 20));
  float* mem_f  = (float*)(arena + ((size_t)8 << 20));

  auto tr = [&](const float* W, bf16* Wt, int K, int N) {
    transpose_bf16_kernel<<<cdiv(K * N, 256), 256, 0, stream>>>(W, Wt, K, N);
  };
  auto gemm = [&](const bf16* A, int lda, const bf16* Bt, int ldb,
                  float* C, int N, int K) {
    gemm_bf16_kernel<<<dim3(N / 256 ? N / 256 : 1, cM / 64), 256, 0, stream>>>(A, lda, Bt, ldb, C, N, K);
  };

  // ---- 0. weight prep ----
  tr(wq, wq_t, 1024, 2048);
  tr(wk, wk_t, 1024, 1024);
  tr(wv, wv_t, 1024, 1024);
  tr(wo, wo_t, 2048, 1024);
  tr(wg, wg_t, 1024, 3072);
  tr(wu, wu_t, 1024, 3072);
  tr(wd, wd_t, 3072, 1024);
  tr(wqr, wqr_t, 1024, 256);
  cvt_f32_bf16_kernel<<<cdiv(2 * 256 * 128, 256), 256, 0, stream>>>(pkk, pk_t, 2 * 256 * 128);
  tr(wfg, wfg_t, 2048, 1024);
  tr(wfm, wfm_t, 1024, 1024);

  // ---- 1. attention ----
  rmsnorm_kernel<<<cM, 256, 0, stream>>>(x_in, ln1, (float*)nullptr, h1_bf, 1024, 1024);
  gemm(h1_bf, 1024, wq_t, 1024, q_f, 2048, 1024);
  gemm(h1_bf, 1024, wk_t, 1024, k_f, 1024, 1024);
  gemm(h1_bf, 1024, wv_t, 1024, v_f, 1024, 1024);
  qknorm_rope_kernel<<<dim3(cM, cH), 128, 0, stream>>>(q_f, qn, q_bf, cH);
  qknorm_rope_kernel<<<dim3(cM, cKV), 128, 0, stream>>>(k_f, kn, k_bf, cKV);
  cvt_f32_bf16_kernel<<<cdiv(cM * 1024, 256), 256, 0, stream>>>(v_f, v_bf, cM * 1024);
  flash_attn_kernel<<<dim3(cS / 16, cH, cB), 32, 0, stream>>>(q_bf, k_bf, v_bf, attn_bf);
  gemm(attn_bf, 2048, wo_t, 2048, oproj, 1024, 2048);
  add_kernel<<<cdiv(cM * 1024, 256), 256, 0, stream>>>(x_in, oproj, x1_f, cM * 1024);

  // ---- 2. MLP ----
  rmsnorm_kernel<<<cM, 256, 0, stream>>>(x1_f, ln2, (float*)nullptr, h2_bf, 1024, 1024);
  gemm(h2_bf, 1024, wg_t, 1024, gate_f, 3072, 1024);
  gemm(h2_bf, 1024, wu_t, 1024, up_f, 3072, 1024);
  silu_mul_kernel<<<cdiv(cM * 3072, 256), 256, 0, stream>>>(gate_f, up_f, act_bf, cM * 3072);
  gemm(act_bf, 3072, wd_t, 3072, mlp_f, 1024, 3072);
  add_kernel<<<cdiv(cM * 1024, 256), 256, 0, stream>>>(x1_f, mlp_f, hs_f, cM * 1024);

  // ---- 3. memory retrieval ----
  rmsnorm_kernel<<<cM, 256, 0, stream>>>(hs_f, memn, hmem_f, cat_bf, 2048, 1024);
  gemm(cat_bf, 2048, wqr_t, 1024, qpk_f, 256, 1024);
  cvt_f32_bf16_kernel<<<cdiv(cM * 256, 256), 256, 0, stream>>>(qpk_f, qpk_bf, cM * 256);
  gemm(qpk_bf, 256, pk_t, 128, s1_f, 256, 128);
  gemm(qpk_bf + 128, 256, pk_t + 256 * 128, 128, s2_f, 256, 128);
  topk_select_kernel<<<cdiv(cM, 128), 128, 0, stream>>>(s1_f, s2_f, idx);
  gather_mem_kernel<<<cM * cNC, 256, 0, stream>>>(idx, bank, temb, mem_f);

  // ---- 4. losses + selection ----
  zero_tail_kernel<<<cdiv(2 + cKNUM, 256), 256, 0, stream>>>(out);
  cos_sim_kernel<<<cM * cNC, 256, 0, stream>>>(hmem_f, mem_f, sim, mnorm);
  div_loss_kernel<<<dim3(cM, 28), 256, 0, stream>>>(mem_f, mnorm, out + DIV_OFF);
  select_gumbel_kernel<<<cdiv(cM, 128), 128, 0, stream>>>(sim, gumbel, idx, selw,
                                                          out + SIM_OFF, out + CNT_OFF);
  selmem_kernel<<<cM, 256, 0, stream>>>(mem_f, selw, cat_bf + 1024);

  // ---- 5. gated fusion ----
  gemm(cat_bf, 2048, wfg_t, 2048, gatelin, 1024, 2048);
  gemm(cat_bf + 1024, 2048, wfm_t, 1024, memlin, 1024, 1024);
  fuse_out_kernel<<<cdiv(cM * 1024, 256), 256, 0, stream>>>(hs_f, gatelin, memlin, out, cM * 1024);
}
